// Attention_15247133900834
// MI455X (gfx1250) — matrix operand order repaired and checked by reading, not compiled, not run on hardware
//
#include <hip/hip_runtime.h>

// ---------------------------------------------------------------------------
// MHA block for MI455X (gfx1250, wave32).
// Pipeline: fp32->bf16 convert, QKV GEMM (WMMA bf16), flash attention with
// post-softmax task mask (WMMA bf16, K/V staged in LDS shared by all waves),
// output projection GEMM (WMMA bf16)+bias.
// All matmuls use v_wmma_f32_16x16x32_bf16 with f32 accumulation.
// ---------------------------------------------------------------------------

typedef __attribute__((ext_vector_type(16))) __bf16 bf16x16;
typedef __attribute__((ext_vector_type(8)))  float  f32x8;

union VecAB { uint4 q[2]; bf16x16 v; };       // 32 bytes: one WMMA A/B operand

__device__ inline unsigned short f2bf(float f) {
    unsigned u = __builtin_bit_cast(unsigned, f);
    u += 0x7FFFu + ((u >> 16) & 1u);          // round-to-nearest-even
    return (unsigned short)(u >> 16);
}

__device__ inline f32x8 zero8() {
    f32x8 z;
#pragma unroll
    for (int i = 0; i < 8; ++i) z[i] = 0.f;
    return z;
}

// ---------------------------------------------------------------------------
// fp32 -> bf16 conversion, 4 elements/thread
// ---------------------------------------------------------------------------
__global__ void cvt_kernel(const float* __restrict__ in,
                           unsigned short* __restrict__ out, int n4) {
    int i = blockIdx.x * blockDim.x + threadIdx.x;
    if (i < n4) {
        float4 f = ((const float4*)in)[i];
        uint2 o;
        o.x = (unsigned)f2bf(f.x) | ((unsigned)f2bf(f.y) << 16);
        o.y = (unsigned)f2bf(f.z) | ((unsigned)f2bf(f.w) << 16);
        ((uint2*)out)[i] = o;
    }
}

// ---------------------------------------------------------------------------
// Generic bf16 GEMM: C[M,Nn] = A[M,K] (row-major bf16) * B[K,Nn] (row-major bf16)
// 256 threads = 8 waves; workgroup tile 128x128, K-step 32.
// Wave (w%4, w/4) computes a 32x64 sub-tile = 2x4 WMMA accumulators.
// mode 0: scatter qkv -> q[b,h,n,d], k[b,h,n,d], vT[b,h,d,n]  (bf16)
// mode 1: out_f32[row*Nn+col] = acc + bias[col]
// ---------------------------------------------------------------------------
#define LDA_S 40   // ushort stride (32 + pad, keeps 16B alignment: 80B rows)
#define LDB_S 40

__global__ __launch_bounds__(256) void gemm_bf16(
    const unsigned short* __restrict__ A,
    const unsigned short* __restrict__ B,
    int M, int Nn, int K, int mode,
    unsigned short* __restrict__ q_out,
    unsigned short* __restrict__ k_out,
    unsigned short* __restrict__ vT_out,
    float* __restrict__ out_f32,
    const float* __restrict__ bias)
{
    __shared__ __align__(16) unsigned short sA[128 * LDA_S]; // [m][k]
    __shared__ __align__(16) unsigned short sB[128 * LDB_S]; // [n][k]  (B^T)

    const int t = threadIdx.x;
    const int wave = t >> 5, lane = t & 31;
    const int lr = lane & 15, lh = lane >> 4;
    const int m0 = blockIdx.y * 128, n0 = blockIdx.x * 128;
    const int wm = (wave & 3) * 32;     // wave M offset in tile
    const int wn = (wave >> 2) * 64;    // wave N offset in tile

    f32x8 acc[2][4];
#pragma unroll
    for (int mi = 0; mi < 2; ++mi)
#pragma unroll
        for (int ni = 0; ni < 4; ++ni) acc[mi][ni] = zero8();

    for (int k0 = 0; k0 < K; k0 += 32) {
        // --- stage A tile: 128 rows x 32 k, 2 lanes/row, 32B each (coalesced)
        {
            int m = t >> 1, hh = (t & 1) * 16;
            const uint4* src = (const uint4*)(A + (size_t)(m0 + m) * K + k0 + hh);
            uint4 a0 = src[0], a1 = src[1];
            uint4* dst = (uint4*)(sA + m * LDA_S + hh);
            dst[0] = a0; dst[1] = a1;
        }
        // --- stage B tile transposed: sB[n][k] so column K-runs are contiguous
        {
            int kk = t & 31, nb = (t >> 5) * 16;
            const uint4* src = (const uint4*)(B + (size_t)(k0 + kk) * Nn + n0 + nb);
            union { uint4 u[2]; unsigned short s[16]; } bb;
            bb.u[0] = src[0]; bb.u[1] = src[1];
#pragma unroll
            for (int j = 0; j < 16; ++j) sB[(nb + j) * LDB_S + kk] = bb.s[j];
        }
        __syncthreads();

#pragma unroll
        for (int mi = 0; mi < 2; ++mi) {
            // A operand layout: lane L holds row (L%16), K run [16h,16h+16)
            VecAB av;
            const unsigned short* pa = sA + (wm + mi * 16 + lr) * LDA_S;
            av.q[0] = *(const uint4*)(pa + 16 * lh);
            av.q[1] = *(const uint4*)(pa + 16 * lh + 8);
#pragma unroll
            for (int ni = 0; ni < 4; ++ni) {
                // B operand layout: lane L holds col (L%16), K run [16h,16h+16)
                VecAB bv;
                const unsigned short* pb = sB + (wn + ni * 16 + lr) * LDB_S + 16 * lh;
                bv.q[0] = *(const uint4*)(pb);
                bv.q[1] = *(const uint4*)(pb + 8);
                acc[mi][ni] = __builtin_amdgcn_wmma_f32_16x16x32_bf16(
                    false, av.v, false, bv.v, (short)0, acc[mi][ni], false, false);
            }
        }
        __syncthreads();
    }

    // --- epilogue.  D layout: VGPR r, lane L -> (row = r + 8*(L/16), col = L%16)
    if (mode == 0) {
#pragma unroll
        for (int mi = 0; mi < 2; ++mi)
#pragma unroll
            for (int ni = 0; ni < 4; ++ni) {
                int col = n0 + wn + ni * 16 + lr;        // [0, 2304)
                int three = col / 768, rem = col % 768;
                int h = rem >> 6, dj = rem & 63;
#pragma unroll
                for (int r = 0; r < 8; ++r) {
                    int row = m0 + wm + mi * 16 + r + 8 * lh;   // [0, 4096)
                    int b = row >> 11, n = row & 2047;
                    unsigned short val = f2bf(acc[mi][ni][r]);
                    size_t bh = (size_t)(b * 12 + h);
                    if (three == 0)      q_out[(bh * 2048 + n) * 64 + dj] = val;
                    else if (three == 1) k_out[(bh * 2048 + n) * 64 + dj] = val;
                    else                 vT_out[(bh * 64 + dj) * 2048 + n] = val;
                }
            }
    } else {
#pragma unroll
        for (int mi = 0; mi < 2; ++mi)
#pragma unroll
            for (int ni = 0; ni < 4; ++ni) {
                int col = n0 + wn + ni * 16 + lr;
                float bv = bias[col];
#pragma unroll
                for (int r = 0; r < 8; ++r) {
                    int row = m0 + wm + mi * 16 + r + 8 * lh;
                    out_f32[(size_t)row * Nn + col] = acc[mi][ni][r] + bv;
                }
            }
    }
}

// ---------------------------------------------------------------------------
// Flash attention, streaming softmax, WMMA bf16.
// grid: (N/128, H, B); 256 threads = 8 waves; each wave owns 16 query rows.
// Key blocks of 64; K and V tiles are staged ONCE per workgroup into LDS
// (8KB each) and shared by all 8 waves -> 8x less L2/HBM traffic than
// per-wave global streaming (the dominant cost on this L2-resident problem).
// Post-softmax task mask: denominator includes masked entries (mask applies
// AFTER softmax in the reference), so we only zero P before the P@V matmul.
// ---------------------------------------------------------------------------
__global__ __launch_bounds__(256) void flash_attn(
    const unsigned short* __restrict__ Q,   // [b,h,n,64] bf16
    const unsigned short* __restrict__ Km,  // [b,h,n,64] bf16
    const unsigned short* __restrict__ Vt,  // [b,h,64,n] bf16 (transposed)
    unsigned short* __restrict__ ctx)       // [b,n,768]  bf16
{
    __shared__ __align__(16) unsigned short sK[64][72];    // [key][d]
    __shared__ __align__(16) unsigned short sV[64][72];    // [dj][key]
    __shared__ __align__(16) unsigned short Pl[8][16][72]; // per-wave P tile

    const int t = threadIdx.x, wave = t >> 5, lane = t & 31;
    const int lr = lane & 15, lh = lane >> 4;
    const int b = blockIdx.z, h = blockIdx.y;
    const int qbase = blockIdx.x * 128 + wave * 16;
    const size_t bh = (size_t)(b * 12 + h);
    const unsigned short* qp = Q  + bh * 2048 * 64;
    const unsigned short* kp = Km + bh * 2048 * 64;
    const unsigned short* vp = Vt + bh * 64 * 2048;

    // Load Q as A operand, K-steps kk=0,1 over d=64
    VecAB aQ[2];
    {
        const unsigned short* qr = qp + (size_t)(qbase + lr) * 64;
#pragma unroll
        for (int kk = 0; kk < 2; ++kk) {
            aQ[kk].q[0] = *(const uint4*)(qr + kk * 32 + 16 * lh);
            aQ[kk].q[1] = *(const uint4*)(qr + kk * 32 + 16 * lh + 8);
        }
    }

    f32x8 O[4];
#pragma unroll
    for (int ni = 0; ni < 4; ++ni) O[ni] = zero8();
    float mrun[8], lrun[8];
#pragma unroll
    for (int r = 0; r < 8; ++r) { mrun[r] = -3.0e38f; lrun[r] = 0.f; }

    const float scale = 0.125f;   // d^-0.5, d=64

    // staging indices: 4 threads per row, 32B (16 ushorts) per thread
    const int srow = t >> 2, sch = (t & 3) * 16;

    for (int j0 = 0; j0 < 2048; j0 += 64) {
        // ---- stage K/V tiles (prior-iteration LDS reads are complete:
        //      their consuming WMMAs precede this barrier in program order)
        __syncthreads();
        {
            const uint4* ks = (const uint4*)(kp + (size_t)(j0 + srow) * 64 + sch);
            uint4 ka = ks[0], kb2 = ks[1];
            const uint4* vs = (const uint4*)(vp + (size_t)srow * 2048 + j0 + sch);
            uint4 va = vs[0], vb = vs[1];
            uint4* kd = (uint4*)(&sK[srow][sch]); kd[0] = ka; kd[1] = kb2;
            uint4* vd = (uint4*)(&sV[srow][sch]); vd[0] = va; vd[1] = vb;
        }
        __syncthreads();

        // ---- S = Q K^T for 16x64 tile (B operand from sK)
        f32x8 S[4];
#pragma unroll
        for (int ni = 0; ni < 4; ++ni) S[ni] = zero8();
#pragma unroll
        for (int ni = 0; ni < 4; ++ni) {
            const unsigned short* kr = &sK[ni * 16 + lr][0];
#pragma unroll
            for (int kk = 0; kk < 2; ++kk) {
                VecAB bK;
                bK.q[0] = *(const uint4*)(kr + kk * 32 + 16 * lh);
                bK.q[1] = *(const uint4*)(kr + kk * 32 + 16 * lh + 8);
                S[ni] = __builtin_amdgcn_wmma_f32_16x16x32_bf16(
                    false, aQ[kk].v, false, bK.v, (short)0, S[ni], false, false);
            }
        }

        // ---- streaming softmax (row = r + 8*lh, reductions over 16-lane half)
        float p[4][8];
#pragma unroll
        for (int r = 0; r < 8; ++r) {
            float mx = fmaxf(fmaxf(S[0][r], S[1][r]), fmaxf(S[2][r], S[3][r]));
            mx = fmaxf(mx, __shfl_xor(mx, 1, 32));
            mx = fmaxf(mx, __shfl_xor(mx, 2, 32));
            mx = fmaxf(mx, __shfl_xor(mx, 4, 32));
            mx = fmaxf(mx, __shfl_xor(mx, 8, 32));
            float mnew  = fmaxf(mrun[r], mx * scale);
            float alpha = __expf(mrun[r] - mnew);
            float rs = 0.f;
#pragma unroll
            for (int ni = 0; ni < 4; ++ni) {
                float e = __expf(S[ni][r] * scale - mnew);
                p[ni][r] = e; rs += e;
            }
            rs += __shfl_xor(rs, 1, 32);
            rs += __shfl_xor(rs, 2, 32);
            rs += __shfl_xor(rs, 4, 32);
            rs += __shfl_xor(rs, 8, 32);
            lrun[r] = lrun[r] * alpha + rs;    // denominator keeps masked terms
            mrun[r] = mnew;
#pragma unroll
            for (int ni = 0; ni < 4; ++ni) O[ni][r] *= alpha;
        }

        // ---- post-softmax task mask (keys 0..3 live in ni==0, block j0==0)
        if (j0 == 0) {
#pragma unroll
            for (int r = 0; r < 8; ++r) {
                int qrow = qbase + r + 8 * lh;
                if (qrow < 4 && lr < 4 && lr != qrow) p[0][r] = 0.f;
            }
        }

        // ---- D-layout -> A-layout via per-wave LDS tile
#pragma unroll
        for (int ni = 0; ni < 4; ++ni)
#pragma unroll
            for (int r = 0; r < 8; ++r)
                Pl[wave][r + 8 * lh][ni * 16 + lr] = f2bf(p[ni][r]);
        __builtin_amdgcn_wave_barrier();   // wave-private tile; DS is in-order

        VecAB aP[2];
#pragma unroll
        for (int kk = 0; kk < 2; ++kk) {
            aP[kk].q[0] = *(const uint4*)(&Pl[wave][lr][kk * 32 + 16 * lh]);
            aP[kk].q[1] = *(const uint4*)(&Pl[wave][lr][kk * 32 + 16 * lh + 8]);
        }

        // ---- O += P V  (B operand from sV: column dj, contiguous over keys)
#pragma unroll
        for (int ni = 0; ni < 4; ++ni) {
            const unsigned short* vr = &sV[ni * 16 + lr][0];
#pragma unroll
            for (int kk = 0; kk < 2; ++kk) {
                VecAB bV;
                bV.q[0] = *(const uint4*)(vr + kk * 32 + 16 * lh);
                bV.q[1] = *(const uint4*)(vr + kk * 32 + 16 * lh + 8);
                O[ni] = __builtin_amdgcn_wmma_f32_16x16x32_bf16(
                    false, aP[kk].v, false, bV.v, (short)0, O[ni], false, false);
            }
        }
        __builtin_amdgcn_wave_barrier();
    }

    // ---- normalize and store ctx[b, n, h*64 + dj]
    float linv[8];
#pragma unroll
    for (int r = 0; r < 8; ++r) linv[r] = 1.f / lrun[r];
#pragma unroll
    for (int ni = 0; ni < 4; ++ni) {
        int c = h * 64 + ni * 16 + lr;
#pragma unroll
        for (int r = 0; r < 8; ++r) {
            int qrow = qbase + r + 8 * lh;
            ctx[((size_t)b * 2048 + qrow) * 768 + c] = f2bf(O[ni][r] * linv[r]);
        }
    }
}

// ---------------------------------------------------------------------------
// Host-side launcher
// ---------------------------------------------------------------------------
extern "C" void kernel_launch(void* const* d_in, const int* in_sizes, int n_in,
                              void* d_out, int out_size, void* d_ws, size_t ws_size,
                              hipStream_t stream) {
    (void)in_sizes; (void)n_in; (void)out_size; (void)ws_size;
    const float* x      = (const float*)d_in[0];   // [2,2048,768]
    const float* w_qkv  = (const float*)d_in[1];   // [768,2304]
    const float* w_proj = (const float*)d_in[2];   // [768,768]
    const float* b_proj = (const float*)d_in[3];   // [768]
    float* out = (float*)d_out;                    // [2,2048,768]

    const int Mtot = 4096, C = 768, C3 = 2304;

    // workspace layout (bf16 = ushort)
    char* ws = (char*)d_ws;
    unsigned short* xbf   = (unsigned short*)(ws);                    // 4096*768
    unsigned short* wqkvb = (unsigned short*)(ws + 6291456);          // 768*2304
    unsigned short* wprjb = (unsigned short*)(ws + 9830400);          // 768*768
    unsigned short* qb    = (unsigned short*)(ws + 11010048);         // 2*12*2048*64
    unsigned short* kb    = (unsigned short*)(ws + 17301504);
    unsigned short* vTb   = (unsigned short*)(ws + 23592960);
    unsigned short* ctxb  = (unsigned short*)(ws + 29884416);         // 4096*768

    // 1) fp32 -> bf16 conversions
    {
        int n4x = Mtot * C / 4, n4q = C * C3 / 4, n4p = C * C / 4;
        cvt_kernel<<<(n4x + 255) / 256, 256, 0, stream>>>(x, xbf, n4x);
        cvt_kernel<<<(n4q + 255) / 256, 256, 0, stream>>>(w_qkv, wqkvb, n4q);
        cvt_kernel<<<(n4p + 255) / 256, 256, 0, stream>>>(w_proj, wprjb, n4p);
    }
    // 2) QKV GEMM -> q/k/vT scatter
    gemm_bf16<<<dim3(C3 / 128, Mtot / 128), 256, 0, stream>>>(
        xbf, wqkvb, Mtot, C3, C, 0, qb, kb, vTb, nullptr, nullptr);
    // 3) flash attention -> ctx
    flash_attn<<<dim3(2048 / 128, 12, 2), 256, 0, stream>>>(qb, kb, vTb, ctxb);
    // 4) projection GEMM + bias -> out (fp32)
    gemm_bf16<<<dim3(C / 128, Mtot / 128), 256, 0, stream>>>(
        ctxb, wprjb, Mtot, C, C, 1, nullptr, nullptr, nullptr, out, b_proj);
}
